// MultiHeadLatentAttention_10359461118450
// MI455X (gfx1250) — matrix operand order repaired and checked by reading, not compiled, not run on hardware
//
#include <hip/hip_runtime.h>

typedef __attribute__((ext_vector_type(16))) __bf16 v16bf;
typedef __attribute__((ext_vector_type(8)))  float  v8f;
typedef __attribute__((ext_vector_type(8)))  int    v8i;
typedef __attribute__((ext_vector_type(4)))  int    v4i;

__device__ __forceinline__ unsigned short bf16bits(float x) {
    return __builtin_bit_cast(unsigned short, (__bf16)x);
}
__device__ __forceinline__ unsigned bf16pair(float a, float b) {
    return (unsigned)bf16bits(a) | ((unsigned)bf16bits(b) << 16);
}
__device__ __forceinline__ float bf2f(unsigned short h) {
    const unsigned u = (unsigned)h << 16;
    return __builtin_bit_cast(float, u);
}
__device__ __forceinline__ v8f vzero() {
    v8f r;
    for (int i = 0; i < 8; ++i) r[i] = 0.f;
    return r;
}

// ---- CDNA5 async global->LDS copy (16B per lane), sync fallback ------------
__device__ __forceinline__ void cp16(void* l, const void* g) {
#if __has_builtin(__builtin_amdgcn_global_load_async_to_lds_b128)
    __builtin_amdgcn_global_load_async_to_lds_b128(
        (__attribute__((address_space(1))) v4i*)g,
        (__attribute__((address_space(3))) v4i*)l, 0, 0);
#else
    *(uint4*)l = *(const uint4*)g;
#endif
}
#if __has_builtin(__builtin_amdgcn_s_wait_asynccnt)
#define WAIT_ASYNC0() __builtin_amdgcn_s_wait_asynccnt(0)
#define WAIT_ASYNC4() __builtin_amdgcn_s_wait_asynccnt(4)
#else
#define WAIT_ASYNC0() asm volatile("s_wait_asynccnt 0x0" ::: "memory")
#define WAIT_ASYNC4() asm volatile("s_wait_asynccnt 0x4" ::: "memory")
#endif

// -------------------------------------------------------------------------
// Elementwise f32 -> bf16 (n multiple of 1024).
// -------------------------------------------------------------------------
__global__ __launch_bounds__(256) void convert_bf16_kernel(
    const float* __restrict__ src, unsigned short* __restrict__ dst)
{
    const size_t i = ((size_t)blockIdx.x * 256 + threadIdx.x) * 4;
    const float4 f = *(const float4*)&src[i];
    uint2 o;
    o.x = bf16pair(f.x, f.y);
    o.y = bf16pair(f.z, f.w);
    *(uint2*)&dst[i] = o;
}

// -------------------------------------------------------------------------
// Transpose W[K,N] f32 -> Wt[N,K] bf16.  Grid (N/32, K/32), 256 threads.
// -------------------------------------------------------------------------
__global__ __launch_bounds__(256) void transpose_bf16_kernel(
    const float* __restrict__ W, unsigned short* __restrict__ Wt, int N, int K)
{
    __shared__ unsigned short t[32][33];
    const int k0 = blockIdx.y * 32;
    const int n0 = blockIdx.x * 32;
    const int tx = threadIdx.x & 31;
    const int ty = threadIdx.x >> 5;  // 0..7
#pragma unroll
    for (int i = 0; i < 4; ++i) {
        const int k = k0 + ty + i * 8;
        t[tx][ty + i * 8] = bf16bits(W[(size_t)k * N + n0 + tx]);
    }
    __syncthreads();
#pragma unroll
    for (int i = 0; i < 4; ++i) {
        const int n = n0 + ty + i * 8;
        Wt[(size_t)n * K + k0 + tx] = t[ty + i * 8][tx];
    }
}

// -------------------------------------------------------------------------
// Per-row rsqrt(mean(x^2)+eps) over bf16 input.
// -------------------------------------------------------------------------
__global__ __launch_bounds__(256) void rowscale_kernel(
    const unsigned short* __restrict__ X, int ldx, int ncols, float inv_n,
    float* __restrict__ out)
{
    __shared__ float red[256];
    const int row = blockIdx.x;
    const unsigned short* xr = X + (size_t)row * ldx;
    float s = 0.f;
    for (int c = threadIdx.x; c < ncols; c += 256) {
        const float v = bf2f(xr[c]);
        s += v * v;
    }
    red[threadIdx.x] = s;
    __syncthreads();
    for (int off = 128; off > 0; off >>= 1) {
        if (threadIdx.x < off) red[threadIdx.x] += red[threadIdx.x + off];
        __syncthreads();
    }
    if (threadIdx.x == 0) out[row] = rsqrtf(red[0] * inv_n + 1e-6f);
}

// -------------------------------------------------------------------------
// dst[m][c] = bf16( src[m][c] * scale[m] * lnw[c] ).  128 thr, 4 cols each.
// Grid (ncols/512, M).
// -------------------------------------------------------------------------
__global__ __launch_bounds__(128) void apply_norm_kernel(
    const unsigned short* __restrict__ src, unsigned short* __restrict__ dst,
    const float* __restrict__ scale, const float* __restrict__ lnw,
    int ld_src, int ld_dst)
{
    const int row = blockIdx.y;
    const int col = (blockIdx.x * 128 + threadIdx.x) * 4;
    const float s = scale[row];
    const uint2 u = *(const uint2*)&src[(size_t)row * ld_src + col];
    const float4 w = *(const float4*)&lnw[col];
    const float f0 = bf2f(u.x & 0xffff) * s * w.x;
    const float f1 = bf2f(u.x >> 16)    * s * w.y;
    const float f2 = bf2f(u.y & 0xffff) * s * w.z;
    const float f3 = bf2f(u.y >> 16)    * s * w.w;
    uint2 o;
    o.x = bf16pair(f0, f1);
    o.y = bf16pair(f2, f3);
    *(uint2*)&dst[(size_t)row * ld_dst + col] = o;
}

// -------------------------------------------------------------------------
// GEMM: C[M,N] = A[M,K](bf16) @ Bt[N,K](bf16)^T.  128x128x32 tiles, 8 waves,
// double-buffered async global->LDS staging (pure copies, no conversion).
// -------------------------------------------------------------------------
template <bool OUT_BF16>
__global__ __launch_bounds__(256) void gemm_bf16_kernel(
    const unsigned short* __restrict__ A, const unsigned short* __restrict__ Bt,
    void* __restrict__ C, int N, int K)
{
    __shared__ unsigned short As[2][128][32];
    __shared__ unsigned short Bs[2][128][32];

    const int bm = blockIdx.y * 128;
    const int bn = blockIdx.x * 128;
    const int tid  = threadIdx.x;
    const int lane = tid & 31;
    const int wv   = tid >> 5;
    const int wm   = (wv >> 1) * 32;
    const int wn   = (wv & 1) * 64;
    const int hf   = lane >> 4;
    const int l16  = lane & 15;

    v8f acc[2][4];
    for (int i = 0; i < 2; ++i)
        for (int j = 0; j < 4; ++j) acc[i][j] = vzero();

    // per-thread staging addresses: 2 x 16B for A, 2 x 16B for B
    const int srow = tid >> 1;            // 0..127
    const int sseg = (tid & 1) * 16;      // bf16 offset 0 / 16

    auto fill = [&](int buf, int k0) {
        cp16(&As[buf][srow][sseg], &A[(size_t)(bm + srow) * K + k0 + sseg]);
        cp16(&Bs[buf][srow][sseg], &Bt[(size_t)(bn + srow) * K + k0 + sseg]);
        cp16(&As[buf][srow][sseg ^ 16], &A[(size_t)(bm + srow) * K + k0 + (sseg ^ 16)]);
        cp16(&Bs[buf][srow][sseg ^ 16], &Bt[(size_t)(bn + srow) * K + k0 + (sseg ^ 16)]);
    };

    fill(0, 0);
    int buf = 0;
    for (int k0 = 0; k0 < K; k0 += 32) {
        const bool more = (k0 + 32) < K;
        if (more) {
            fill(buf ^ 1, k0 + 32);
            WAIT_ASYNC4();
        } else {
            WAIT_ASYNC0();
        }
        __syncthreads();

        v16bf af[2], bfm[4];
#pragma unroll
        for (int mt = 0; mt < 2; ++mt) {
            v8i t;
            const int m = wm + mt * 16 + l16;
#pragma unroll
            for (int v = 0; v < 8; ++v) {
                const int k = hf * 8 + (v & 3) * 2 + (v >> 2) * 16;
                t[v] = *(const unsigned*)&As[buf][m][k];
            }
            af[mt] = __builtin_bit_cast(v16bf, t);
        }
#pragma unroll
        for (int nt = 0; nt < 4; ++nt) {
            v8i t;
            const int n = wn + nt * 16 + l16;
#pragma unroll
            for (int v = 0; v < 8; ++v) {
                const int k = hf * 8 + (v & 3) * 2 + (v >> 2) * 16;
                t[v] = *(const unsigned*)&Bs[buf][n][k];
            }
            bfm[nt] = __builtin_bit_cast(v16bf, t);
        }
#pragma unroll
        for (int mt = 0; mt < 2; ++mt)
#pragma unroll
            for (int nt = 0; nt < 4; ++nt)
                acc[mt][nt] = __builtin_amdgcn_wmma_f32_16x16x32_bf16(
                    false, af[mt], false, bfm[nt], (short)0, acc[mt][nt], false, false);
        __syncthreads();
        buf ^= 1;
    }

#pragma unroll
    for (int mt = 0; mt < 2; ++mt)
#pragma unroll
        for (int nt = 0; nt < 4; ++nt)
#pragma unroll
            for (int r = 0; r < 8; ++r) {
                const size_t gm = bm + wm + mt * 16 + r + 8 * hf;
                const size_t gn = bn + wn + nt * 16 + l16;
                if constexpr (OUT_BF16)
                    ((unsigned short*)C)[gm * N + gn] = bf16bits(acc[mt][nt][r]);
                else
                    ((float*)C)[gm * N + gn] = acc[mt][nt][r];
            }
}

// -------------------------------------------------------------------------
// Flash attention for MLA (bf16 in, bf16 out). Grid (T/64, H, B), 128 thr.
//   q:      [4096,3072] bf16, head h at cols h*192
//   kv:     [4096,4096] bf16, head h: k_nope at h*256, v at h*256+128
//   kv_raw: [4096,1536] bf16, rope at 512 + h*64
//   attn:   [4096,2048] bf16, head h at cols h*128
// -------------------------------------------------------------------------
__global__ __launch_bounds__(128) void mla_flash_kernel(
    const unsigned short* __restrict__ q, const unsigned short* __restrict__ kv,
    const unsigned short* __restrict__ kv_raw, unsigned short* __restrict__ attn)
{
    __shared__ unsigned short Ks[32][192];
    __shared__ unsigned short Vs[128][32];
    __shared__ unsigned short Ps[4][16][32];

    const int b  = blockIdx.z;
    const int h  = blockIdx.y;
    const int q0 = blockIdx.x * 64;
    const int tid  = threadIdx.x;
    const int lane = tid & 31;
    const int wv   = tid >> 5;
    const int hf   = lane >> 4;
    const int l16  = lane & 15;
    const float sc = 0.07216878364870322f;  // 1/sqrt(192)

    // ---- Q fragments: straight bf16-pair loads ----
    v16bf qf[6];
    {
        const int qrow = q0 + wv * 16 + l16;
        const unsigned short* qp = q + (size_t)(b * 2048 + qrow) * 3072 + h * 192;
#pragma unroll
        for (int s = 0; s < 6; ++s) {
            v8i t;
#pragma unroll
            for (int v = 0; v < 8; ++v) {
                const int k = s * 32 + hf * 8 + (v & 3) * 2 + (v >> 2) * 16;
                t[v] = *(const unsigned*)&qp[k];
            }
            qf[s] = __builtin_bit_cast(v16bf, t);
        }
    }

    v8f acc[8];
    for (int i = 0; i < 8; ++i) acc[i] = vzero();
    float mrow[8], lrow[8];
    for (int r = 0; r < 8; ++r) { mrow[r] = -1e38f; lrow[r] = 0.f; }

    const int jend = q0 + 64;
    for (int j0 = 0; j0 < jend; j0 += 32) {
        // ---- stage K tile (async pure copies) + V tile (transposed scatter) ----
        {
            const int key  = tid >> 2;
            const int krow = b * 2048 + j0 + key;
            const unsigned short* nope = kv + (size_t)krow * 4096 + h * 256;
            const unsigned short* rope = kv_raw + (size_t)krow * 1536 + 512 + h * 64;
            const int d0 = (tid & 3) * 48;
#pragma unroll
            for (int c = 0; c < 6; ++c) {
                const int d = d0 + c * 8;
                const unsigned short* src = (d < 128) ? (nope + d) : (rope + d - 128);
                cp16(&Ks[key][d], src);
            }
            const unsigned short* vp = kv + (size_t)krow * 4096 + h * 256 + 128;
            const int vd0 = (tid & 3) * 32;
#pragma unroll
            for (int c = 0; c < 4; ++c) {
                const uint4 u = *(const uint4*)&vp[vd0 + c * 8];
                const unsigned uu[4] = {u.x, u.y, u.z, u.w};
#pragma unroll
                for (int e = 0; e < 4; ++e) {
                    Vs[vd0 + c * 8 + e * 2][key]     = (unsigned short)(uu[e] & 0xffff);
                    Vs[vd0 + c * 8 + e * 2 + 1][key] = (unsigned short)(uu[e] >> 16);
                }
            }
        }
        WAIT_ASYNC0();
        __syncthreads();

        // ---- scores: two 16x16 tiles over d=192 ----
        v8f c[2];
#pragma unroll
        for (int nt = 0; nt < 2; ++nt) {
            c[nt] = vzero();
#pragma unroll
            for (int s = 0; s < 6; ++s) {
                v8i t;
#pragma unroll
                for (int v = 0; v < 8; ++v) {
                    const int k = s * 32 + hf * 8 + (v & 3) * 2 + (v >> 2) * 16;
                    t[v] = *(const unsigned*)&Ks[nt * 16 + l16][k];
                }
                const v16bf kf = __builtin_bit_cast(v16bf, t);
                c[nt] = __builtin_amdgcn_wmma_f32_16x16x32_bf16(
                    false, qf[s], false, kf, (short)0, c[nt], false, false);
            }
#pragma unroll
            for (int r = 0; r < 8; ++r) c[nt][r] *= sc;
        }

        // ---- causal mask + online softmax ----
        const int qbase = q0 + wv * 16 + 8 * hf;
#pragma unroll
        for (int nt = 0; nt < 2; ++nt) {
            const int kcol = j0 + nt * 16 + l16;
#pragma unroll
            for (int r = 0; r < 8; ++r)
                if (kcol > qbase + r) c[nt][r] = -1e30f;
        }
#pragma unroll
        for (int r = 0; r < 8; ++r) {
            float rm = fmaxf(c[0][r], c[1][r]);
            rm = fmaxf(rm, __shfl_xor(rm, 1));
            rm = fmaxf(rm, __shfl_xor(rm, 2));
            rm = fmaxf(rm, __shfl_xor(rm, 4));
            rm = fmaxf(rm, __shfl_xor(rm, 8));
            const float mnew = fmaxf(mrow[r], rm);
            const float corr = __expf(mrow[r] - mnew);
            const float p0 = __expf(c[0][r] - mnew);
            const float p1 = __expf(c[1][r] - mnew);
            float ps = p0 + p1;
            ps += __shfl_xor(ps, 1);
            ps += __shfl_xor(ps, 2);
            ps += __shfl_xor(ps, 4);
            ps += __shfl_xor(ps, 8);
            lrow[r] = lrow[r] * corr + ps;
            mrow[r] = mnew;
            c[0][r] = p0;
            c[1][r] = p1;
#pragma unroll
            for (int vt = 0; vt < 8; ++vt) acc[vt][r] *= corr;
        }

        // ---- probs C-layout -> A-layout via per-wave LDS patch ----
#pragma unroll
        for (int nt = 0; nt < 2; ++nt)
#pragma unroll
            for (int r = 0; r < 8; ++r)
                Ps[wv][r + 8 * hf][nt * 16 + l16] = bf16bits(c[nt][r]);

        v8i pt;
#pragma unroll
        for (int v = 0; v < 8; ++v) {
            const int k = hf * 8 + (v & 3) * 2 + (v >> 2) * 16;
            pt[v] = *(const unsigned*)&Ps[wv][l16][k];
        }
        const v16bf pf = __builtin_bit_cast(v16bf, pt);

        // ---- O += P @ V ----
#pragma unroll
        for (int vt = 0; vt < 8; ++vt) {
            v8i t;
#pragma unroll
            for (int v = 0; v < 8; ++v) {
                const int k = hf * 8 + (v & 3) * 2 + (v >> 2) * 16;
                t[v] = *(const unsigned*)&Vs[vt * 16 + l16][k];
            }
            const v16bf vf = __builtin_bit_cast(v16bf, t);
            acc[vt] = __builtin_amdgcn_wmma_f32_16x16x32_bf16(
                false, pf, false, vf, (short)0, acc[vt], false, false);
        }
        __syncthreads();
    }

    // ---- normalize + store bf16 ----
#pragma unroll
    for (int vt = 0; vt < 8; ++vt)
#pragma unroll
        for (int r = 0; r < 8; ++r) {
            const int m = r + 8 * hf;
            const size_t grow = b * 2048 + q0 + wv * 16 + m;
            attn[grow * 2048 + h * 128 + vt * 16 + l16] = bf16bits(acc[vt][r] / lrow[r]);
        }
}

extern "C" void kernel_launch(void* const* d_in, const int* in_sizes, int n_in,
                              void* d_out, int out_size, void* d_ws, size_t ws_size,
                              hipStream_t stream) {
    (void)in_sizes; (void)n_in; (void)out_size; (void)ws_size;
    const float* hidden  = (const float*)d_in[0];
    const float* q_a_w   = (const float*)d_in[1];
    const float* q_a_ln  = (const float*)d_in[2];
    const float* q_b_w   = (const float*)d_in[3];
    const float* kv_a_w  = (const float*)d_in[4];
    const float* kv_a_ln = (const float*)d_in[5];
    const float* kv_b_w  = (const float*)d_in[6];
    const float* o_w     = (const float*)d_in[7];
    float* out = (float*)d_out;

    const int M = 4096;  // B*T
    unsigned short* ws = (unsigned short*)d_ws;
    unsigned short* hidden_bf = ws;                             // 4096*2048
    unsigned short* wq_a_t    = hidden_bf + (size_t)M * 2048;   // 1536*2048
    unsigned short* wkv_a_t   = wq_a_t  + (size_t)1536 * 2048;  // 1536*2048
    unsigned short* wq_b_t    = wkv_a_t + (size_t)1536 * 2048;  // 3072*1536
    unsigned short* wkv_b_t   = wq_b_t  + (size_t)3072 * 1536;  // 4096*512
    unsigned short* wo_t      = wkv_b_t + (size_t)4096 * 512;   // 2048*2048
    unsigned short* q_a_out   = wo_t    + (size_t)2048 * 2048;  // 4096*1536
    unsigned short* kv_raw    = q_a_out + (size_t)M * 1536;     // 4096*1536
    unsigned short* q_a_n     = kv_raw  + (size_t)M * 1536;     // 4096*1536
    unsigned short* kv_n      = q_a_n   + (size_t)M * 1536;     // 4096*512
    unsigned short* qbuf      = kv_n    + (size_t)M * 512;      // 4096*3072
    unsigned short* kvbuf     = qbuf    + (size_t)M * 3072;     // 4096*4096
    unsigned short* attn      = kvbuf   + (size_t)M * 4096;     // 4096*2048
    float* q_scale  = (float*)(attn + (size_t)M * 2048);        // 4096
    float* kv_scale = q_scale + M;                              // 4096

    // ---- one-time conversions (per call; bandwidth-trivial) ----
    convert_bf16_kernel<<<(size_t)M * 2048 / 1024, 256, 0, stream>>>(hidden, hidden_bf);
    transpose_bf16_kernel<<<dim3(1536 / 32, 2048 / 32), 256, 0, stream>>>(q_a_w, wq_a_t, 1536, 2048);
    transpose_bf16_kernel<<<dim3(1536 / 32, 2048 / 32), 256, 0, stream>>>(kv_a_w, wkv_a_t, 1536, 2048);
    transpose_bf16_kernel<<<dim3(3072 / 32, 1536 / 32), 256, 0, stream>>>(q_b_w, wq_b_t, 3072, 1536);
    transpose_bf16_kernel<<<dim3(4096 / 32, 512 / 32), 256, 0, stream>>>(kv_b_w, wkv_b_t, 4096, 512);
    transpose_bf16_kernel<<<dim3(2048 / 32, 2048 / 32), 256, 0, stream>>>(o_w, wo_t, 2048, 2048);

    // ---- low-rank projections ----
    gemm_bf16_kernel<true><<<dim3(1536 / 128, M / 128), 256, 0, stream>>>(
        hidden_bf, wq_a_t, q_a_out, 1536, 2048);
    gemm_bf16_kernel<true><<<dim3(1536 / 128, M / 128), 256, 0, stream>>>(
        hidden_bf, wkv_a_t, kv_raw, 1536, 2048);

    // ---- rmsnorm ----
    rowscale_kernel<<<M, 256, 0, stream>>>(q_a_out, 1536, 1536, 1.f / 1536.f, q_scale);
    rowscale_kernel<<<M, 256, 0, stream>>>(kv_raw, 1536, 512, 1.f / 512.f, kv_scale);
    apply_norm_kernel<<<dim3(1536 / 512, M), 128, 0, stream>>>(
        q_a_out, q_a_n, q_scale, q_a_ln, 1536, 1536);
    apply_norm_kernel<<<dim3(512 / 512, M), 128, 0, stream>>>(
        kv_raw, kv_n, kv_scale, kv_a_ln, 1536, 512);

    // ---- up projections ----
    gemm_bf16_kernel<true><<<dim3(3072 / 128, M / 128), 256, 0, stream>>>(
        q_a_n, wq_b_t, qbuf, 3072, 1536);
    gemm_bf16_kernel<true><<<dim3(4096 / 128, M / 128), 256, 0, stream>>>(
        kv_n, wkv_b_t, kvbuf, 4096, 512);

    // ---- attention ----
    mla_flash_kernel<<<dim3(2048 / 64, 16, 2), 128, 0, stream>>>(qbuf, kvbuf, kv_raw, attn);

    // ---- output projection (f32 out) ----
    gemm_bf16_kernel<false><<<dim3(2048 / 128, M / 128), 256, 0, stream>>>(
        attn, wo_t, out, 2048, 2048);
}
